// NeuronRouter_71167608094941
// MI455X (gfx1250) — compile-verified, matrix operand order checked by reading, not run on hardware
//
#include <hip/hip_runtime.h>
#include <cstdint>
#include <cstddef>

#define B_  8
#define S_  1024
#define D_  1024
#define H_  16
#define DH_ 64
#define N_  1024
#define R_  32
#define K_  16
#define M_  (B_*S_)   // 8192 tokens

typedef __attribute__((ext_vector_type(4)))  __bf16 v4bf;
typedef __attribute__((ext_vector_type(8)))  __bf16 v8bf;
typedef __attribute__((ext_vector_type(16))) __bf16 v16bf;
typedef __attribute__((ext_vector_type(8)))  float  v8f;
typedef uint32_t u32x4 __attribute__((ext_vector_type(4)));
typedef int      i32x4 __attribute__((ext_vector_type(4)));
typedef int      i32x8 __attribute__((ext_vector_type(8)));

#if defined(__has_builtin)
#if __has_builtin(__builtin_amdgcn_tensor_load_to_lds)
#define HAVE_TDM 1
#endif
#endif
#ifndef HAVE_TDM
#define HAVE_TDM 0
#endif

// therock-10.0 headers (6-arg TDM builtin) ship amd_gfx1250_TDM.h; ROCm 7.2 does not.
#if __has_include(<hip/amd_detail/amd_gfx1250_TDM.h>)
#define TDM_ARGS6 1
#else
#define TDM_ARGS6 0
#endif

static __device__ inline __bf16 f2bf(float f) { return (__bf16)f; }   // native v_cvt

// ---------------------------------------------------------------------------
// Fragment loaders.  A-fragment (16x32 bf16) layout per CDNA5 ISA 7.12.2:
//   lane l: row = l%16; lanes 0-15 hold K=0..7 then 16..23, lanes 16-31 hold
//   K=8..15 then 24..31.  B fragments mirrored with n = l%16, so B is stored
//   (n,k)-major.  Each fragment = two contiguous 16-byte loads per lane.
// ---------------------------------------------------------------------------
static __device__ inline v16bf load_frag(const __bf16* base, int ld, int row0, int k0) {
  int lane = threadIdx.x & 31;
  int r  = row0 + (lane & 15);
  int hi = lane >> 4;
  const __bf16* p = base + (size_t)r * ld + k0 + 8 * hi;
  v8bf lo = *(const v8bf*)(p);
  v8bf hp = *(const v8bf*)(p + 16);
  v16bf out;
#pragma unroll
  for (int i = 0; i < 8; i++) { out[i] = lo[i]; out[i + 8] = hp[i]; }
  return out;
}

static __device__ inline v16bf load_frag_f32(const float* base, int ld, int row0, int k0) {
  int lane = threadIdx.x & 31;
  int r  = row0 + (lane & 15);
  int hi = lane >> 4;
  const float* p = base + (size_t)r * ld + k0 + 8 * hi;
  v16bf out;
#pragma unroll
  for (int i = 0; i < 8; i++) { out[i] = f2bf(p[i]); out[i + 8] = f2bf(p[i + 16]); }
  return out;
}

static __device__ inline v8f vzero() {
  v8f z;
#pragma unroll
  for (int i = 0; i < 8; i++) z[i] = 0.f;
  return z;
}

#define WMMA_BF16(a, b, c) \
  __builtin_amdgcn_wmma_f32_16x16x32_bf16(false, (a), false, (b), (short)0, (c), false, false)

// ---------------------------------------------------------------------------
// TDM: DMA a 2D tile (tile_k contiguous bf16 x tile_rows rows, row stride in
// elems) from global into LDS at byte offset lds_off.  D# per ISA 8.3/8.4.
// Wave-uniform; issued by one wave; ordered with s_wait_tensorcnt.
// ---------------------------------------------------------------------------
#if HAVE_TDM
static __device__ __attribute__((always_inline)) void
tdm_load_2d_bf16(uint32_t lds_off, const __bf16* gptr, uint32_t tile_k,
                 uint32_t tile_rows, uint32_t row_stride_elems) {
  uint64_t ga = (uint64_t)(uintptr_t)gptr;
  u32x4 g0;
  g0[0] = 1u;                                            // count=1, user desc
  g0[1] = lds_off;                                       // lds_addr (bytes)
  g0[2] = (uint32_t)ga;                                  // global_addr[31:0]
  g0[3] = (uint32_t)((ga >> 32) & 0x01FFFFFFu) | 0x80000000u;  // addr[56:32], type=2
  uint32_t td0 = row_stride_elems;                       // tensor_dim0 >= tile span
  uint32_t td1 = 1u << 20;                               // tensor_dim1: large
  i32x8 g1;
  g1[0] = (int)(1u << 16);                               // wg_mask=0, data_size=1 (2B)
  g1[1] = (int)((td0 & 0xFFFFu) << 16);                  // tensor_dim0 lo16
  g1[2] = (int)((td0 >> 16) | ((td1 & 0xFFFFu) << 16));  // dim0 hi16 | dim1 lo16
  g1[3] = (int)((td1 >> 16) | (tile_k << 16));           // dim1 hi16 | tile_dim0
  g1[4] = (int)tile_rows;                                // tile_dim1 | tile_dim2=0
  g1[5] = (int)row_stride_elems;                         // tensor_dim0_stride lo32
  g1[6] = 0;                                             // stride hi | dim1_stride lo
  g1[7] = 0;
  i32x4 z4 = {0, 0, 0, 0};
#if TDM_ARGS6
  i32x8 z8 = {0, 0, 0, 0, 0, 0, 0, 0};
  __builtin_amdgcn_tensor_load_to_lds(g0, g1, z4, z4, z8, 0);
#else
  __builtin_amdgcn_tensor_load_to_lds(g0, g1, z4, z4, 0);
#endif
}
#endif

// ---------------------------------------------------------------------------
// 1) f32 -> bf16 convert (vectorized: float4 in, 8-byte bf16x4 out).
// ---------------------------------------------------------------------------
__global__ __launch_bounds__(256) void k_cvt_bf16(const float* __restrict__ in,
                                                  __bf16* __restrict__ out, int n) {
  int i = (blockIdx.x * 256 + threadIdx.x) * 4;
  if (i < n) {
    float4 f = *(const float4*)(in + i);
    v4bf o; o[0] = f2bf(f.x); o[1] = f2bf(f.y); o[2] = f2bf(f.z); o[3] = f2bf(f.w);
    *(v4bf*)(out + i) = o;
  }
}

// neuron_B (R,D) -> nBt (D,R) bf16  ((n,k)-major for the compose GEMM)
__global__ __launch_bounds__(256) void k_nbt(const float* __restrict__ nB,
                                             __bf16* __restrict__ nBt) {
  int i = blockIdx.x * 256 + threadIdx.x;           // i = d*R + r
  if (i < D_ * R_) {
    int d = i / R_, r = i % R_;
    nBt[i] = f2bf(nB[r * D_ + d]);
  }
}

// ---------------------------------------------------------------------------
// 2) neurons = neuron_A @ neuron_B  (1024x1024, K=32): one WMMA per tile.
// ---------------------------------------------------------------------------
__global__ __launch_bounds__(256) void k_compose(const float* __restrict__ nA,
                                                 const __bf16* __restrict__ nBt,
                                                 float* __restrict__ neurons,
                                                 __bf16* __restrict__ neurons_bf) {
  int wave = threadIdx.x >> 5, lane = threadIdx.x & 31;
  int tile = blockIdx.x * 8 + wave;                  // 4096 tiles
  int tm = (tile >> 6) * 16;                         // neuron row
  int tn = (tile & 63) * 16;                         // d col
  v16bf a = load_frag_f32(nA, R_, tm, 0);
  v16bf b = load_frag(nBt, R_, tn, 0);
  v8f c = WMMA_BF16(a, b, vzero());
  int nl = lane & 15, hi = lane >> 4;
#pragma unroll
  for (int v = 0; v < 8; v++) {
    int row = tm + v + 8 * hi, col = tn + nl;
    size_t off = (size_t)row * D_ + col;
    neurons[off] = c[v];
    neurons_bf[off] = f2bf(c[v]);
  }
}

// ---------------------------------------------------------------------------
// 3) Generic bf16 WMMA GEMM, TDM-staged double-buffered 64x32 B tiles in LDS.
//    Wave tile 32x64 (8 acc tiles, 8 WMMA per K-step); WG = 8 waves = 256 M.
//    MODE 0: bf16 head layout (b,h,s,dh);  MODE 2: (b,h,dh,s);  MODE 3: f32.
// ---------------------------------------------------------------------------
template <int MODE>
__global__ __launch_bounds__(256) void k_gemm(const __bf16* __restrict__ A,
                                              const __bf16* __restrict__ Bm,
                                              const float* __restrict__ bias,
                                              __bf16* __restrict__ outb,
                                              float* __restrict__ outf,
                                              int Kdim, int Nn) {
  __shared__ __align__(16) __bf16 Bt[2][64 * 32];    // sole LDS object -> offset 0
  int wave = threadIdx.x >> 5, lane = threadIdx.x & 31;
  int mtile = blockIdx.y * 256 + wave * 32;
  int nbase = blockIdx.x * 64;
  v8f acc[2][4];
#pragma unroll
  for (int mt = 0; mt < 2; mt++)
#pragma unroll
    for (int t = 0; t < 4; t++) acc[mt][t] = vzero();

  const int nsteps = Kdim >> 5;
  const __bf16* Brow = Bm + (size_t)nbase * Kdim;

#if HAVE_TDM
  if (wave == 0)
    tdm_load_2d_bf16(0u, Brow, 32u, 64u, (uint32_t)Kdim);
  for (int i = 0; i < nsteps; i++) {
    if (wave == 0) {
      if (i + 1 < nsteps) {
        tdm_load_2d_bf16((uint32_t)(((i + 1) & 1) * 64 * 32 * 2),
                         Brow + (size_t)(i + 1) * 32, 32u, 64u, (uint32_t)Kdim);
        __builtin_amdgcn_s_wait_tensorcnt(1);
      } else {
        __builtin_amdgcn_s_wait_tensorcnt(0);
      }
    }
    __syncthreads();                                  // tile i visible to all waves
    __builtin_prefetch(A + (size_t)mtile * Kdim + (size_t)(i + 2) * 32, 0, 0);
    v16bf a0 = load_frag(A, Kdim, mtile, i * 32);
    v16bf a1 = load_frag(A, Kdim, mtile + 16, i * 32);
    const __bf16* bt = Bt[i & 1];
#pragma unroll
    for (int t = 0; t < 4; t++) {
      v16bf b = load_frag(bt, 32, 16 * t, 0);
      acc[0][t] = WMMA_BF16(a0, b, acc[0][t]);
      acc[1][t] = WMMA_BF16(a1, b, acc[1][t]);
    }
    __syncthreads();                                  // done reading before rewrite
  }
#else
  for (int i = 0; i < nsteps; i++) {
    v16bf a0 = load_frag(A, Kdim, mtile, i * 32);
    v16bf a1 = load_frag(A, Kdim, mtile + 16, i * 32);
#pragma unroll
    for (int t = 0; t < 4; t++) {
      v16bf b = load_frag(Bm, Kdim, nbase + 16 * t, i * 32);
      acc[0][t] = WMMA_BF16(a0, b, acc[0][t]);
      acc[1][t] = WMMA_BF16(a1, b, acc[1][t]);
    }
  }
#endif

  int nl = lane & 15, hi = lane >> 4;
#pragma unroll
  for (int mt = 0; mt < 2; mt++) {
#pragma unroll
    for (int t = 0; t < 4; t++) {
#pragma unroll
      for (int v = 0; v < 8; v++) {
        int m = mtile + 16 * mt + v + 8 * hi;
        int n = nbase + 16 * t + nl;
        float val = acc[mt][t][v] + (bias ? bias[n] : 0.f);
        if (MODE == 3) {
          outf[(size_t)m * Nn + n] = val;
        } else {
          int b_ = m >> 10, s = m & (S_ - 1);
          int h = n >> 6, dh = n & 63;
          size_t off = (MODE == 2)
              ? (((size_t)(b_ * H_ + h)) * DH_ + dh) * S_ + s
              : (((size_t)(b_ * H_ + h)) * S_ + s) * DH_ + dh;
          outb[off] = f2bf(val);
        }
      }
    }
  }
}

// ---------------------------------------------------------------------------
// 4) Flash attention.  One wave = 16 query rows; block = 8 waves = 128 rows,
//    all sharing one (b,h).  K (32x64) and V (64x32) tiles are TDM-DMA'd into
//    LDS once per block, double-buffered on TENSORcnt (8x less global traffic
//    than per-wave loads).  Online softmax via shfl row-reductions; P tile
//    bounced through wave-private LDS (C-layout -> A-layout).
//    LDS map (bytes): K[2] @ 0/4096, V[2] @ 8192/12288, P-stage @ 16384.
// ---------------------------------------------------------------------------
__global__ __launch_bounds__(256) void k_attn(const __bf16* __restrict__ qh,
                                              const __bf16* __restrict__ kh,
                                              const __bf16* __restrict__ vt,
                                              float* __restrict__ ctx_out,
                                              __bf16* __restrict__ ctx_bf) {
  __shared__ __align__(16) __bf16 smem[12288];       // sole LDS object -> offset 0
  __bf16* kst = smem;                                // 2 x 2048 (32 rows x 64)
  __bf16* vst = smem + 4096;                         // 2 x 2048 (64 rows x 32)
  int wave = threadIdx.x >> 5, lane = threadIdx.x & 31;
  __bf16* pw = smem + 8192 + wave * 512;             // 16 x 32 per wave

  int bh = blockIdx.x >> 3;
  int rowBase = (blockIdx.x & 7) * 128 + wave * 16;
  const __bf16* q  = qh + (size_t)bh * S_ * DH_;
  const __bf16* kp = kh + (size_t)bh * S_ * DH_;
  const __bf16* vp = vt + (size_t)bh * DH_ * S_;

  v16bf qa0 = load_frag(q, DH_, rowBase, 0);
  v16bf qa1 = load_frag(q, DH_, rowBase, 32);

  float mrun[8], lrun[8];
  v8f o[4];
#pragma unroll
  for (int t = 0; t < 4; t++) o[t] = vzero();
#pragma unroll
  for (int v = 0; v < 8; v++) { mrun[v] = -__builtin_inff(); lrun[v] = 0.f; }

  int nl = lane & 15, hi = lane >> 4;
  const int NSTEP = S_ / 32;

#if HAVE_TDM
  if (wave == 0) {
    tdm_load_2d_bf16(0u,    kp, 64u, 32u, (uint32_t)DH_);   // K tile 0
    tdm_load_2d_bf16(8192u, vp, 32u, 64u, (uint32_t)S_);    // V tile 0
  }
#endif
  for (int i = 0; i < NSTEP; i++) {
    int t0 = i * 32;
#if HAVE_TDM
    if (wave == 0) {
      if (i + 1 < NSTEP) {
        uint32_t buf = (uint32_t)((i + 1) & 1) * 4096u;
        tdm_load_2d_bf16(buf,         kp + (size_t)(t0 + 32) * DH_, 64u, 32u, (uint32_t)DH_);
        tdm_load_2d_bf16(8192u + buf, vp + (t0 + 32),               32u, 64u, (uint32_t)S_);
        __builtin_amdgcn_s_wait_tensorcnt(2);       // previous pair complete
      } else {
        __builtin_amdgcn_s_wait_tensorcnt(0);
      }
    }
    __syncthreads();
    const __bf16* kbuf = kst + (i & 1) * 2048;
    const __bf16* vbuf = vst + (i & 1) * 2048;
#else
    __syncthreads();
    {   // cooperative staging fallback
      const __bf16* kg = kp + (size_t)t0 * DH_;     // 32x64 contiguous
      for (int e = threadIdx.x * 8; e < 2048; e += 256 * 8)
        *(v8bf*)(kst + e) = *(const v8bf*)(kg + e);
      int r = threadIdx.x >> 2, c = (threadIdx.x & 3) * 8;
      *(v8bf*)(vst + r * 32 + c) = *(const v8bf*)(vp + (size_t)r * S_ + t0 + c);
    }
    __syncthreads();
    const __bf16* kbuf = kst;
    const __bf16* vbuf = vst;
#endif
    v16bf b00 = load_frag(kbuf, DH_, 0, 0);
    v16bf b01 = load_frag(kbuf, DH_, 0, 32);
    v16bf b10 = load_frag(kbuf, DH_, 16, 0);
    v16bf b11 = load_frag(kbuf, DH_, 16, 32);
    v8f s0 = WMMA_BF16(qa0, b00, WMMA_BF16(qa1, b01, vzero()));
    v8f s1 = WMMA_BF16(qa0, b10, WMMA_BF16(qa1, b11, vzero()));

#pragma unroll
    for (int v = 0; v < 8; v++) {
      float a = s0[v] * 0.125f, b = s1[v] * 0.125f;    // 1/sqrt(64)
      float mx = fmaxf(a, b);
#pragma unroll
      for (int msk = 8; msk; msk >>= 1) mx = fmaxf(mx, __shfl_xor(mx, msk, 32));
      float mnew = fmaxf(mrun[v], mx);
      float scaleO = __expf(mrun[v] - mnew);
      float p0 = __expf(a - mnew), p1 = __expf(b - mnew);
      s0[v] = p0; s1[v] = p1;
      float rs = p0 + p1;
#pragma unroll
      for (int msk = 8; msk; msk >>= 1) rs += __shfl_xor(rs, msk, 32);
      lrun[v] = lrun[v] * scaleO + rs;
      mrun[v] = mnew;
#pragma unroll
      for (int t = 0; t < 4; t++) o[t][v] *= scaleO;
    }

    // stage P (16x32) to LDS in C-layout, reload as A fragment
#pragma unroll
    for (int v = 0; v < 8; v++) {
      int m = v + 8 * hi;
      pw[m * 32 + nl] = f2bf(s0[v]);
      pw[m * 32 + 16 + nl] = f2bf(s1[v]);
    }
    asm volatile("s_wait_dscnt 0" ::: "memory");
    v16bf pf = load_frag(pw, 32, 0, 0);

#pragma unroll
    for (int t = 0; t < 4; t++) {
      v16bf vb = load_frag(vbuf, 32, t * 16, 0);   // rows = dh, cols = t-local
      o[t] = WMMA_BF16(pf, vb, o[t]);
    }
    __syncthreads();                               // done with tiles before rewrite
  }

  int b_ = bh / H_, h = bh % H_;
#pragma unroll
  for (int t = 0; t < 4; t++) {
#pragma unroll
    for (int v = 0; v < 8; v++) {
      int m = v + 8 * hi;
      int s = rowBase + m;
      int col = h * 64 + t * 16 + nl;
      float val = o[t][v] / lrun[v];
      size_t off = ((size_t)(b_ * S_ + s)) * D_ + col;
      ctx_out[off] = val;
      ctx_bf[off] = f2bf(val);
    }
  }
}

// ---------------------------------------------------------------------------
// 5) Path weights + mix:  w = softmax([x,ctx] @ path_w^T + b);
//    mixed = bf16(w0*x + w1*ctx).  One token per block.
// ---------------------------------------------------------------------------
__global__ __launch_bounds__(256) void k_path(const float* __restrict__ x,
                                              const float* __restrict__ ctx,
                                              const float* __restrict__ pw,
                                              const float* __restrict__ pb,
                                              __bf16* __restrict__ mixed) {
  __shared__ float red0[256], red1[256];
  int tok = blockIdx.x;
  const float* xr = x + (size_t)tok * D_;
  const float* cr = ctx + (size_t)tok * D_;
  float s0 = 0.f, s1 = 0.f;
  for (int d = threadIdx.x; d < D_; d += 256) {
    float xv = xr[d], cv = cr[d];
    s0 += xv * pw[d] + cv * pw[D_ + d];
    s1 += xv * pw[2 * D_ + d] + cv * pw[3 * D_ + d];
  }
  red0[threadIdx.x] = s0; red1[threadIdx.x] = s1;
  __syncthreads();
  for (int off = 128; off; off >>= 1) {
    if (threadIdx.x < off) {
      red0[threadIdx.x] += red0[threadIdx.x + off];
      red1[threadIdx.x] += red1[threadIdx.x + off];
    }
    __syncthreads();
  }
  float l0 = red0[0] + pb[0], l1 = red1[0] + pb[1];
  float mx = fmaxf(l0, l1);
  float e0 = __expf(l0 - mx), e1 = __expf(l1 - mx);
  float w0 = e0 / (e0 + e1), w1 = e1 / (e0 + e1);
  for (int d = threadIdx.x; d < D_; d += 256)
    mixed[(size_t)tok * D_ + d] = f2bf(w0 * xr[d] + w1 * cr[d]);
}

// ---------------------------------------------------------------------------
// 6) Top-K (16 of 1024) per token + softmax of selected scores.
// ---------------------------------------------------------------------------
__global__ __launch_bounds__(256) void k_topk(const float* __restrict__ scores,
                                              int* __restrict__ idx_out,
                                              float* __restrict__ wt_out) {
  __shared__ float cv[8][512];
  __shared__ int   ci[8][512];
  int wave = threadIdx.x >> 5, lane = threadIdx.x & 31;
  int row = blockIdx.x * 8 + wave;
  const float* sr = scores + (size_t)row * N_;

  float lv[K_]; int li[K_];
#pragma unroll
  for (int i = 0; i < K_; i++) { lv[i] = -__builtin_inff(); li[i] = 0; }
  for (int j = lane; j < N_; j += 32) {
    float v = sr[j];
    if (v > lv[K_ - 1]) {
      int p = K_ - 1;
      while (p > 0 && lv[p - 1] < v) { lv[p] = lv[p - 1]; li[p] = li[p - 1]; p--; }
      lv[p] = v; li[p] = j;
    }
  }
#pragma unroll
  for (int i = 0; i < K_; i++) { cv[wave][lane * K_ + i] = lv[i]; ci[wave][lane * K_ + i] = li[i]; }
  asm volatile("s_wait_dscnt 0" ::: "memory");

  float selv[K_]; int seli[K_];
  for (int j = 0; j < K_; j++) {
    float bv = -__builtin_inff(); int bslot = 0;
#pragma unroll
    for (int i = 0; i < K_; i++) {
      float v = cv[wave][lane * K_ + i];
      if (v > bv) { bv = v; bslot = lane * K_ + i; }
    }
#pragma unroll
    for (int msk = 16; msk; msk >>= 1) {
      float ov = __shfl_xor(bv, msk, 32);
      int os = __shfl_xor(bslot, msk, 32);
      if (ov > bv || (ov == bv && os < bslot)) { bv = ov; bslot = os; }
    }
    selv[j] = bv;
    seli[j] = ci[wave][bslot];
    if ((bslot >> 4) == lane) cv[wave][bslot] = -__builtin_inff();
    asm volatile("s_wait_dscnt 0" ::: "memory");
  }

  float sum = 0.f;
#pragma unroll
  for (int j = 0; j < K_; j++) sum += __expf(selv[j] - selv[0]);
  if (lane < K_) {
    idx_out[(size_t)row * K_ + lane] = seli[lane];
    wt_out[(size_t)row * K_ + lane] = __expf(selv[lane] - selv[0]) / sum;
  }
}

// ---------------------------------------------------------------------------
// 7) Gather: selected[b,s,k,:] = neurons[idx] (512 MB streaming float4 copy).
// ---------------------------------------------------------------------------
__global__ __launch_bounds__(256) void k_gather(const int* __restrict__ idx,
                                                const float* __restrict__ neurons,
                                                float* __restrict__ sel) {
  int tok = blockIdx.x;
#pragma unroll
  for (int kk = 0; kk < K_; kk++) {
    int nid = idx[(size_t)tok * K_ + kk];
    const float4* src = (const float4*)(neurons + (size_t)nid * D_);
    float4* dst = (float4*)(sel + ((size_t)tok * K_ + kk) * D_);
    dst[threadIdx.x] = src[threadIdx.x];             // 256 * float4 = 1024 f32
  }
}

// ---------------------------------------------------------------------------
extern "C" void kernel_launch(void* const* d_in, const int* in_sizes, int n_in,
                              void* d_out, int out_size, void* d_ws, size_t ws_size,
                              hipStream_t stream) {
  const float* x  = (const float*)d_in[0];
  const float* nA = (const float*)d_in[1];
  const float* nB = (const float*)d_in[2];
  const float* qw = (const float*)d_in[3];
  const float* qb = (const float*)d_in[4];
  const float* kw = (const float*)d_in[5];
  const float* kb = (const float*)d_in[6];
  const float* vw = (const float*)d_in[7];
  const float* vb = (const float*)d_in[8];
  const float* pw = (const float*)d_in[9];
  const float* pb = (const float*)d_in[10];

  const size_t SEL = (size_t)M_ * K_ * D_;     // 134217728
  const size_t IDX = (size_t)M_ * K_;          // 131072
  float* out  = (float*)d_out;
  float* sel  = out;
  int*   idxo = (int*)(out + SEL);
  float* wto  = out + SEL + IDX;
  float* ctxo = out + SEL + 2 * IDX;           // (B,S,D) f32 context

  char* ws = (char*)d_ws;
  size_t off = 0;
  auto alloc = [&](size_t bytes) -> void* {
    void* p = ws + off; off += (bytes + 255) & ~(size_t)255; return p;
  };
  __bf16* xbf   = (__bf16*)alloc((size_t)M_ * D_ * 2);
  __bf16* qwbf  = (__bf16*)alloc((size_t)D_ * D_ * 2);
  __bf16* kwbf  = (__bf16*)alloc((size_t)D_ * D_ * 2);
  __bf16* vwbf  = (__bf16*)alloc((size_t)D_ * D_ * 2);
  __bf16* nBt   = (__bf16*)alloc((size_t)D_ * R_ * 2);
  float*  neur  = (float*) alloc((size_t)N_ * D_ * 4);
  __bf16* neurb = (__bf16*)alloc((size_t)N_ * D_ * 2);
  __bf16* qh    = (__bf16*)alloc((size_t)B_ * H_ * S_ * DH_ * 2);
  __bf16* kh    = (__bf16*)alloc((size_t)B_ * H_ * S_ * DH_ * 2);
  __bf16* vt    = (__bf16*)alloc((size_t)B_ * H_ * DH_ * S_ * 2);
  __bf16* ctxb  = (__bf16*)alloc((size_t)M_ * D_ * 2);
  __bf16* mixed = (__bf16*)alloc((size_t)M_ * D_ * 2);
  float*  scor  = (float*) alloc((size_t)M_ * N_ * 4);
  (void)ws_size; (void)in_sizes; (void)n_in; (void)out_size; (void)ctxb;

  k_cvt_bf16<<<(M_ * D_ / 4 + 255) / 256, 256, 0, stream>>>(x, xbf, M_ * D_);
  k_cvt_bf16<<<(D_ * D_ / 4 + 255) / 256, 256, 0, stream>>>(qw, qwbf, D_ * D_);
  k_cvt_bf16<<<(D_ * D_ / 4 + 255) / 256, 256, 0, stream>>>(kw, kwbf, D_ * D_);
  k_cvt_bf16<<<(D_ * D_ / 4 + 255) / 256, 256, 0, stream>>>(vw, vwbf, D_ * D_);
  k_nbt<<<(D_ * R_ + 255) / 256, 256, 0, stream>>>(nB, nBt);

  k_compose<<<512, 256, 0, stream>>>(nA, nBt, neur, neurb);

  dim3 ggrid(N_ / 64, M_ / 256);
  k_gemm<0><<<ggrid, 256, 0, stream>>>(xbf, qwbf, qb, qh, nullptr, D_, D_);
  k_gemm<0><<<ggrid, 256, 0, stream>>>(xbf, kwbf, kb, kh, nullptr, D_, D_);
  k_gemm<2><<<ggrid, 256, 0, stream>>>(xbf, vwbf, vb, vt, nullptr, D_, D_);

  k_attn<<<B_ * H_ * (S_ / 128), 256, 0, stream>>>(qh, kh, vt, ctxo, ctxb);

  k_path<<<M_, 256, 0, stream>>>(x, ctxo, pw, pb, mixed);

  k_gemm<3><<<ggrid, 256, 0, stream>>>(mixed, neurb, nullptr, nullptr, scor, D_, N_);

  k_topk<<<M_ / 8, 256, 0, stream>>>(scor, idxo, wto);

  k_gather<<<M_, 256, 0, stream>>>(idxo, neur, sel);
}
